// Actor_att_55456617726454
// MI455X (gfx1250) — matrix operand order repaired and checked by reading, not compiled, not run on hardware
//
#include <hip/hip_runtime.h>
#include <stdint.h>

// ---------------- types ----------------
typedef __attribute__((ext_vector_type(16))) _Float16 v16h;
typedef __attribute__((ext_vector_type(8)))  _Float16 h8;
typedef __attribute__((ext_vector_type(2)))  _Float16 h2v;
typedef __attribute__((ext_vector_type(8)))  float    v8f;
typedef __attribute__((ext_vector_type(4)))  float    f4;
typedef __attribute__((ext_vector_type(2)))  float    f2;

union U16h { v16h v; _Float16 h[16]; };
union U8f  { v8f  v; float    f[8];  };

#define EPSF 1e-5f

struct Params {
  const float *x;
  const float *Ws1,*bs1,*Ws2,*bs2;
  const float *Wf1,*bf1,*Wf2,*bf2,*gf,*betaf;
  const float *Wo1,*bo1,*Wo2,*bo2,*go,*betao;
  const float *Wm1,*bm1,*Wm2,*bm2,*Wm3,*bm3;
  float *out;
  int B, tiles;
};

// ---------------- helpers ----------------
// K index owned by this lane-half at A-fragment slot e (16-bit A 16x32 layout):
// e in [0,8): K = 8*half + e ; e in [8,16): K = 16 + 8*half + (e-8)
static __device__ inline int kofe(int e, int half) {
  return e + 8 * half + ((e >= 8) ? 8 : 0);
}

// XOR lane swap via ds_swizzle immediate (group-of-32: and=0x1F, or=0, xor=M)
template <int M>
static __device__ inline float xsw(float v) {
  int r = __builtin_amdgcn_ds_swizzle(__builtin_bit_cast(int, v),
                                      (M << 10) | 0x1F);
  return __builtin_bit_cast(float, r);
}
// sum across the 16 lanes of this half (xor masks < 16 never cross halves)
static __device__ inline float hred16(float v) {
  v += xsw<1>(v); v += xsw<2>(v); v += xsw<4>(v); v += xsw<8>(v);
  return v;
}

static __device__ inline v8f wmma16(v16h a, v16h b, v8f c) {
  return __builtin_amdgcn_wmma_f32_16x16x32_f16(
      /*neg_a=*/false, a, /*neg_b=*/false, b,
      /*c_mod=*/(short)0, c, /*reuse_a=*/false, /*reuse_b=*/false);
}

static __device__ inline float lrelu(float x) { return fmaxf(x, 0.01f * x); }

// B fragment (KxN tile, 16-bit B layout): lane ln holds column `col`;
// slot e holds K=koff+kofe(e,half) (same K order as the A fragment).
static __device__ inline v16h load_bfrag(const float* W, int cols, int col,
                                         int koff, int krows, int half) {
  U16h b;
#pragma unroll
  for (int e = 0; e < 16; ++e) {
    int k = koff + kofe(e, half);
    b.h[e] = (_Float16)((k < krows) ? W[k * cols + col] : 0.0f);
  }
  return b.v;
}

// First layer (2 -> 32) computed straight into A-fragment slots for this lane.
static __device__ inline v16h layer1(const h2v* w01, const h2v* bp,
                                     float x0, float x1) {
  U16h a;
#pragma unroll
  for (int e = 0; e < 16; ++e) {
    float w0 = (float)w01[e].x;
    float w1 = (float)w01[e].y;
    float b  = (float)((e & 1) ? bp[e >> 1].y : bp[e >> 1].x);
    float h  = fmaf(x0, w0, fmaf(x1, w1, b));
    a.h[e]   = (_Float16)fmaxf(h, 0.0f);
  }
  return a.v;
}

// Attention branch: NE entities -> kv via WMMA, dot with self_out, softmax,
// pool, LayerNorm, ReLU. Everything lives in the WMMA D layout.
// Score/softmax/LN fused per accumulator row r to minimize live ranges.
template <int NE>
static __device__ inline void branch(const h2v* w01, const h2v* bp, v16h Bf,
                                     float b2_l, float g_l, float be_l,
                                     const float* ex, const float* ey,
                                     const U8f& self, U8f& outp) {
  U8f kv[NE];
#pragma unroll
  for (int ent = 0; ent < NE; ++ent) {
    v16h a = layer1(w01, bp, ex[ent], ey[ent]);
    v8f c = {};
    kv[ent].v = wmma16(a, Bf, c);
#pragma unroll
    for (int r = 0; r < 8; ++r)
      kv[ent].f[r] = fmaxf(kv[ent].f[r] + b2_l, 0.0f);
  }
#pragma unroll
  for (int r = 0; r < 8; ++r) {
    float s[NE];
#pragma unroll
    for (int ent = 0; ent < NE; ++ent)
      s[ent] = hred16(self.f[r] * kv[ent].f[r]) * 0.25f; // 1/sqrt(16)
    float mx = s[0];
#pragma unroll
    for (int ent = 1; ent < NE; ++ent) mx = fmaxf(mx, s[ent]);
    float es[NE], ssum = 0.0f;
#pragma unroll
    for (int ent = 0; ent < NE; ++ent) { es[ent] = __expf(s[ent] - mx); ssum += es[ent]; }
    float inv = 1.0f / ssum;
    float pl = 0.0f;
#pragma unroll
    for (int ent = 0; ent < NE; ++ent) pl = fmaf(es[ent] * inv, kv[ent].f[r], pl);
    // LayerNorm over the 16 features (lanes of this half)
    float s1 = hred16(pl);
    float s2 = hred16(pl * pl);
    float mean = s1 * 0.0625f;
    float var  = s2 * 0.0625f - mean * mean;
    float nz   = (pl - mean) * rsqrtf(var + EPSF);
    outp.f[r]  = fmaxf(fmaf(g_l, nz, be_l), 0.0f);
  }
}

// ---------------- kernel ----------------
__global__ void __launch_bounds__(256, 4) actor_att_kernel(Params p) {
  const int lane = threadIdx.x & 31;
  const int w    = threadIdx.x >> 5;
  const int half = lane >> 4;
  const int ln   = lane & 15;

  __shared__ __align__(16) _Float16 ltile[8][16][64];
  // Merge-layer B fragments are identical for every wave (lane-dependent
  // only): build once in wave 0, reload transiently per tile (saves 48
  // persistent VGPRs per lane).
  // [0..1]=Wm1 k0 ntile{0,1}, [2..3]=Wm1 k1 ntile{0,1}, [4..5]=Wm2 ntile{0,1}
  __shared__ __align__(32) _Float16 wfrag[6][32][16];

  if (w == 0) {
#pragma unroll
    for (int u = 0; u < 2; ++u) {
      v16h t0 = load_bfrag(p.Wm1, 32, 16 * u + ln, 0,  48, half);
      v16h t1 = load_bfrag(p.Wm1, 32, 16 * u + ln, 32, 48, half); // K>=48 padded 0
      v16h t2 = load_bfrag(p.Wm2, 32, 16 * u + ln, 0,  32, half);
      *(v16h*)&wfrag[0 + u][lane][0] = t0;
      *(v16h*)&wfrag[2 + u][lane][0] = t1;
      *(v16h*)&wfrag[4 + u][lane][0] = t2;
    }
  }
  // zero the K=48..63 pad columns once (B frag is zero there too; this just
  // guards against NaN * 0)
#pragma unroll
  for (int r = 0; r < 8; ++r) ltile[w][r + 8 * half][48 + ln] = (_Float16)0.0f;

  // ---- persistent per-lane weights (loop invariant) ----
  h2v w01s[16], w01f[16], w01o[16];
  h2v bps[8], bpf[8], bpo[8];
#pragma unroll
  for (int e = 0; e < 16; ++e) {
    int k = kofe(e, half);
    w01s[e].x = (_Float16)p.Ws1[k]; w01s[e].y = (_Float16)p.Ws1[32 + k];
    w01f[e].x = (_Float16)p.Wf1[k]; w01f[e].y = (_Float16)p.Wf1[32 + k];
    w01o[e].x = (_Float16)p.Wo1[k]; w01o[e].y = (_Float16)p.Wo1[32 + k];
  }
#pragma unroll
  for (int i = 0; i < 8; ++i) {
    int k0 = kofe(2 * i, half), k1 = kofe(2 * i + 1, half);
    bps[i].x = (_Float16)p.bs1[k0]; bps[i].y = (_Float16)p.bs1[k1];
    bpf[i].x = (_Float16)p.bf1[k0]; bpf[i].y = (_Float16)p.bf1[k1];
    bpo[i].x = (_Float16)p.bo1[k0]; bpo[i].y = (_Float16)p.bo1[k1];
  }

  v16h Bs2 = load_bfrag(p.Ws2, 16, ln, 0, 32, half);
  v16h Bf2 = load_bfrag(p.Wf2, 16, ln, 0, 32, half);
  v16h Bo2 = load_bfrag(p.Wo2, 16, ln, 0, 32, half);

  const float bs2_l = p.bs2[ln], bf2_l = p.bf2[ln], bo2_l = p.bo2[ln];
  const float gf_l = p.gf[ln], bef_l = p.betaf[ln];
  const float go_l = p.go[ln], beo_l = p.betao[ln];
  const float bm1_l[2] = {p.bm1[ln], p.bm1[16 + ln]};
  const float bm2_l[2] = {p.bm2[ln], p.bm2[16 + ln]};
  const float w3a0 = p.Wm3[2 * ln],        w3a1 = p.Wm3[2 * ln + 1];
  const float w3b0 = p.Wm3[2 * (16 + ln)], w3b1 = p.Wm3[2 * (16 + ln) + 1];
  const float b30 = p.bm3[0], b31 = p.bm3[1];

  __syncthreads(); // wfrag ready for all waves

  const int gw     = blockIdx.x * 8 + w;
  const int stride = gridDim.x * 8;

  for (int t = gw; t < p.tiles; t += stride) {
    const int m0 = t * 16;
    int mi = m0 + ln; if (mi >= p.B) mi = p.B - 1; // clamp, keep EXEC full
    const float* xp = p.x + (size_t)mi * 12;
    f4 xa = *(const f4*)(xp);
    f4 xb = *(const f4*)(xp + 4);
    f4 xc = *(const f4*)(xp + 8);

    // ---- self branch: 2->32 (VALU, A layout) then 32->16 via WMMA ----
    v16h aS = layer1(w01s, bps, xa.x, xa.y);
    U8f self; { v8f c = {}; self.v = wmma16(aS, Bs2, c); }
#pragma unroll
    for (int r = 0; r < 8; ++r) self.f[r] = fmaxf(self.f[r] + bs2_l, 0.0f);

    // ---- food / other attention branches (3 + 2 WMMAs) ----
    U8f foodo, othero;
    { float ex[3] = {xa.z, xb.x, xb.z}, ey[3] = {xa.w, xb.y, xb.w};
      branch<3>(w01f, bpf, Bf2, bf2_l, gf_l, bef_l, ex, ey, self, foodo); }
    { float ex[2] = {xc.x, xc.z}, ey[2] = {xc.y, xc.w};
      branch<2>(w01o, bpo, Bo2, bo2_l, go_l, beo_l, ex, ey, self, othero); }

    // ---- D layout -> LDS (16x48 merged tile, f16) ----
#pragma unroll
    for (int r = 0; r < 8; ++r) {
      int m = r + 8 * half;
      ltile[w][m][ln]      = (_Float16)self.f[r];
      ltile[w][m][16 + ln] = (_Float16)foodo.f[r];
      ltile[w][m][32 + ln] = (_Float16)othero.f[r];
    }
    asm volatile("s_wait_dscnt 0x0" ::: "memory"); // same-wave transpose fence

    // ---- read back in A layout: row=element ln, contiguous K runs ----
    const _Float16* row = &ltile[w][ln][0];
    h8 lo0 = *(const h8*)(row + 8 * half);
    h8 hi0 = *(const h8*)(row + 16 + 8 * half);
    h8 lo1 = *(const h8*)(row + 32 + 8 * half);
    h8 hi1 = *(const h8*)(row + 48 + 8 * half);
    v16h a0 = __builtin_shufflevector(lo0, hi0, 0,1,2,3,4,5,6,7,8,9,10,11,12,13,14,15);
    v16h a1 = __builtin_shufflevector(lo1, hi1, 0,1,2,3,4,5,6,7,8,9,10,11,12,13,14,15);

    // ---- merged 48->32 : 4 WMMAs (K 32 + 16-padded, two N tiles) ----
    U8f h2o[2];
#pragma unroll
    for (int u = 0; u < 2; ++u) {
      v16h bk0 = *(const v16h*)&wfrag[0 + u][lane][0];
      v16h bk1 = *(const v16h*)&wfrag[2 + u][lane][0];
      v8f c = {};
      c = wmma16(a0, bk0, c);
      c = wmma16(a1, bk1, c);
      U8f acc; acc.v = c;
#pragma unroll
      for (int r = 0; r < 8; ++r) h2o[u].f[r] = lrelu(acc.f[r] + bm1_l[u]);
    }

    // ---- second transpose for 32->32 ----
#pragma unroll
    for (int u = 0; u < 2; ++u)
#pragma unroll
      for (int r = 0; r < 8; ++r)
        ltile[w][r + 8 * half][16 * u + ln] = (_Float16)h2o[u].f[r];
    asm volatile("s_wait_dscnt 0x0" ::: "memory");

    h8 lo2 = *(const h8*)(row + 8 * half);
    h8 hi2 = *(const h8*)(row + 16 + 8 * half);
    v16h a2 = __builtin_shufflevector(lo2, hi2, 0,1,2,3,4,5,6,7,8,9,10,11,12,13,14,15);

    // ---- 32->32 : 2 WMMAs ----
    U8f h3o[2];
#pragma unroll
    for (int u = 0; u < 2; ++u) {
      v16h b2 = *(const v16h*)&wfrag[4 + u][lane][0];
      v8f c = {};
      U8f acc; acc.v = wmma16(a2, b2, c);
#pragma unroll
      for (int r = 0; r < 8; ++r) h3o[u].f[r] = lrelu(acc.f[r] + bm2_l[u]);
    }

    // ---- 32->2 : lane reduction over the feature dim ----
    float p0[8], p1[8];
#pragma unroll
    for (int r = 0; r < 8; ++r) {
      p0[r] = hred16(fmaf(h3o[0].f[r], w3a0, h3o[1].f[r] * w3b0)) + b30;
      p1[r] = hred16(fmaf(h3o[0].f[r], w3a1, h3o[1].f[r] * w3b1)) + b31;
    }
    if (ln == 0) {
#pragma unroll
      for (int r = 0; r < 8; ++r) {
        int m_el = m0 + r + 8 * half;
        if (m_el < p.B) {
          f2 o; o.x = p0[r]; o.y = p1[r];
          *(f2*)(p.out + 2 * (size_t)m_el) = o;
        }
      }
    }
  }
}

// ---------------- host ----------------
extern "C" void kernel_launch(void* const* d_in, const int* in_sizes, int n_in,
                              void* d_out, int out_size, void* d_ws, size_t ws_size,
                              hipStream_t stream) {
  (void)n_in; (void)out_size; (void)d_ws; (void)ws_size;
  Params p;
  p.x    = (const float*)d_in[0];
  p.Ws1  = (const float*)d_in[1];  p.bs1  = (const float*)d_in[2];
  p.Ws2  = (const float*)d_in[3];  p.bs2  = (const float*)d_in[4];
  p.Wf1  = (const float*)d_in[5];  p.bf1  = (const float*)d_in[6];
  p.Wf2  = (const float*)d_in[7];  p.bf2  = (const float*)d_in[8];
  p.gf   = (const float*)d_in[9];  p.betaf= (const float*)d_in[10];
  p.Wo1  = (const float*)d_in[11]; p.bo1  = (const float*)d_in[12];
  p.Wo2  = (const float*)d_in[13]; p.bo2  = (const float*)d_in[14];
  p.go   = (const float*)d_in[15]; p.betao= (const float*)d_in[16];
  p.Wm1  = (const float*)d_in[17]; p.bm1  = (const float*)d_in[18];
  p.Wm2  = (const float*)d_in[19]; p.bm2  = (const float*)d_in[20];
  p.Wm3  = (const float*)d_in[21]; p.bm3  = (const float*)d_in[22];
  p.out  = (float*)d_out;
  p.B    = in_sizes[0] / 12;
  p.tiles = (p.B + 15) / 16;

  int blocks = (p.tiles + 7) / 8;
  if (blocks > 1024) blocks = 1024;
  if (blocks < 1) blocks = 1;
  actor_att_kernel<<<dim3(blocks), dim3(256), 0, stream>>>(p);
}
